// Encoder_54657753809483
// MI455X (gfx1250) — matrix operand order repaired and checked by reading, not compiled, not run on hardware
//
#include <hip/hip_runtime.h>
#include <hip/hip_bf16.h>

typedef __attribute__((ext_vector_type(16))) _Float16 v16h;
typedef __attribute__((ext_vector_type(8)))  _Float16 v8h;
typedef __attribute__((ext_vector_type(8)))  float    v8f;

#define BB   8192
#define TT   256
#define HH   512
#define H3   1536
#define MT   4          // M-tiles (16 rows each) per wave

__device__ __forceinline__ float sigmoidf_(float x) {
    return 1.0f / (1.0f + __expf(-x));
}
__device__ __forceinline__ float tanhf_(float x) {
    return 2.0f * sigmoidf_(2.0f * x) - 1.0f;
}
__device__ __forceinline__ v16h pack16(v8h lo, v8h hi) {
    v16h r;
#pragma unroll
    for (int t = 0; t < 8; ++t) { r[t] = lo[t]; r[8 + t] = hi[t]; }
    return r;
}

struct Frag { v8h lo, hi; };

// A fragment (16-bit A 16x32 layout): lanes 0-15 K{kb..+7, kb+16..+23},
// lanes 16-31 K{kb+8..+15, kb+24..+31}; pointer pre-offset by hi16*8.
__device__ __forceinline__ void load_a(Frag& f, const _Float16* p, int kb) {
    f.lo = *(const v8h*)(p + kb);
    f.hi = *(const v8h*)(p + kb + 16);
}
// B fragment (16-bit B 32x16 layout): lanes 0-15 K{kb..+7, kb+16..+23},
// lanes 16-31 K{kb+8..+15, kb+24..+31}; pointer pre-offset by hi16*8.
__device__ __forceinline__ void load_b(Frag& f, const _Float16* p, int kb) {
    f.lo = *(const v8h*)(p + kb);
    f.hi = *(const v8h*)(p + kb + 16);
}

// ---------------------------------------------------------------------------
// One-time per launch: R16t[n*H + k] = (f16) R[k*3H + n]   (n in [0,3H))
// ---------------------------------------------------------------------------
__global__ void enc_prep_R(const float* __restrict__ R, _Float16* __restrict__ R16t) {
    int idx = blockIdx.x * 256 + threadIdx.x;          // over 3H*H
    int n = idx >> 9;                                  // /512
    int k = idx & 511;
    R16t[idx] = (_Float16)R[(size_t)k * H3 + n];
}

// v[n] = sum_k dk[k]*K[k,n] ; c[n] = sum_k db[k]*K[k,n] + b0[n]
__global__ void enc_prep_vc(const float* __restrict__ dk, const float* __restrict__ db,
                            const float* __restrict__ Kk, const float* __restrict__ gb0,
                            float* __restrict__ vvec, float* __restrict__ cvec) {
    int n = blockIdx.x * 256 + threadIdx.x;
    if (n >= H3) return;
    float sv = 0.f, sc = 0.f;
    for (int k = 0; k < HH; ++k) {
        float w = Kk[(size_t)k * H3 + n];
        sv = fmaf(dk[k], w, sv);
        sc = fmaf(db[k], w, sc);
    }
    vvec[n] = sv;
    cvec[n] = sc + gb0[n];
}

// ---------------------------------------------------------------------------
// Step 0: h_prev = 0 so mh = b1.  Pure elementwise.
// ---------------------------------------------------------------------------
__global__ void enc_step0(const float* __restrict__ x, const float* __restrict__ km,
                          const float* __restrict__ vvec, const float* __restrict__ cvec,
                          const float* __restrict__ gb1,
                          float* __restrict__ h32, _Float16* __restrict__ h16) {
    int idx = blockIdx.x * 256 + threadIdx.x;          // over B*H
    int b = idx >> 9;
    int n = idx & 511;
    float col = x[(size_t)b * TT] / km[0];
    float xz = fmaf(col, vvec[n],          cvec[n]);
    float xr = fmaf(col, vvec[n + HH],     cvec[n + HH]);
    float xh = fmaf(col, vvec[n + 2*HH],   cvec[n + 2*HH]);
    float z  = sigmoidf_(xz + gb1[n]);
    float r  = sigmoidf_(xr + gb1[n + HH]);
    float hh = tanhf_(xh + r * gb1[n + 2*HH]);
    float h  = (1.0f - z) * hh;
    h32[idx] = h;
    h16[idx] = (_Float16)h;
}

// ---------------------------------------------------------------------------
// Steps 1..T-1: fused  mh = h @ R (f16 WMMA, f32 acc)  +  rank-1 mx  +  gates
// grid = (B/512, H/16), block = 256 (8 waves; wave w -> 64 rows at m0)
// Each wave: 4 M-tiles x 3 gate N-tiles = 12 WMMA accumulators.
// Explicit two-phase ping-pong over 64-wide K super-steps: no register
// rotation copies, loads always overlap WMMA execution.
// ---------------------------------------------------------------------------
__global__ __launch_bounds__(256) void enc_step(
    const float*    __restrict__ h_in32,
    const _Float16* __restrict__ h_in16,
    const _Float16* __restrict__ R16t,     // [3H][H], row n = column n of R
    const float*    __restrict__ gb1,      // 3H
    const float*    __restrict__ vvec,     // 3H
    const float*    __restrict__ cvec,     // 3H
    const float*    __restrict__ km, int step,
    float*          __restrict__ h_out32,
    _Float16*       __restrict__ h_out16) {

    const int lane = threadIdx.x & 31;
    const int wave = threadIdx.x >> 5;
    const int m0 = blockIdx.x * (16 * MT * 8) + wave * (16 * MT);
    const int j0 = blockIdx.y * 16;
    const int l15  = lane & 15;
    const int hi16 = (lane >> 4) & 1;

    v8f acc[MT][3];
#pragma unroll
    for (int t = 0; t < MT; ++t)
#pragma unroll
        for (int g = 0; g < 3; ++g) acc[t][g] = (v8f){};

    const _Float16* aptr[MT];
#pragma unroll
    for (int t = 0; t < MT; ++t)
        aptr[t] = h_in16 + (size_t)(m0 + t * 16 + l15) * HH + hi16 * 8;
    const _Float16* bptr[3];
#pragma unroll
    for (int g = 0; g < 3; ++g)
        bptr[g] = R16t + (size_t)(j0 + l15 + g * HH) * HH + hi16 * 8;

    Frag A0[MT], B0[3], A1[MT], B1[3];
#pragma unroll
    for (int t = 0; t < MT; ++t) load_a(A0[t], aptr[t], 0);
#pragma unroll
    for (int g = 0; g < 3; ++g) load_b(B0[g], bptr[g], 0);

#pragma unroll 2
    for (int kb = 0; kb < HH; kb += 64) {
        // ---- prefetch phase-1 operands (kb+32 always < HH; HH % 64 == 0) ----
#pragma unroll
        for (int t = 0; t < MT; ++t) load_a(A1[t], aptr[t], kb + 32);
#pragma unroll
        for (int g = 0; g < 3; ++g) load_b(B1[g], bptr[g], kb + 32);

        // ---- compute with phase-0 ----
        {
            v16h bv0 = pack16(B0[0].lo, B0[0].hi);
            v16h bv1 = pack16(B0[1].lo, B0[1].hi);
            v16h bv2 = pack16(B0[2].lo, B0[2].hi);
#pragma unroll
            for (int t = 0; t < MT; ++t) {
                v16h av = pack16(A0[t].lo, A0[t].hi);
                acc[t][0] = __builtin_amdgcn_wmma_f32_16x16x32_f16(false, av, false, bv0, (short)0, acc[t][0], false, false);
                acc[t][1] = __builtin_amdgcn_wmma_f32_16x16x32_f16(false, av, false, bv1, (short)0, acc[t][1], false, false);
                acc[t][2] = __builtin_amdgcn_wmma_f32_16x16x32_f16(false, av, false, bv2, (short)0, acc[t][2], false, false);
            }
        }

        // ---- prefetch next phase-0 operands ----
        if (kb + 64 < HH) {
#pragma unroll
            for (int t = 0; t < MT; ++t) load_a(A0[t], aptr[t], kb + 64);
#pragma unroll
            for (int g = 0; g < 3; ++g) load_b(B0[g], bptr[g], kb + 64);
        }

        // ---- compute with phase-1 ----
        {
            v16h bv0 = pack16(B1[0].lo, B1[0].hi);
            v16h bv1 = pack16(B1[1].lo, B1[1].hi);
            v16h bv2 = pack16(B1[2].lo, B1[2].hi);
#pragma unroll
            for (int t = 0; t < MT; ++t) {
                v16h av = pack16(A1[t].lo, A1[t].hi);
                acc[t][0] = __builtin_amdgcn_wmma_f32_16x16x32_f16(false, av, false, bv0, (short)0, acc[t][0], false, false);
                acc[t][1] = __builtin_amdgcn_wmma_f32_16x16x32_f16(false, av, false, bv1, (short)0, acc[t][1], false, false);
                acc[t][2] = __builtin_amdgcn_wmma_f32_16x16x32_f16(false, av, false, bv2, (short)0, acc[t][2], false, false);
            }
        }
    }

    // Gate math. C/D layout: VGPR e -> M = e + 8*hi16, lane l15 -> N.
    const int n = j0 + l15;
    const float invkm = 1.0f / km[step];
    const float gz = gb1[n], gr = gb1[n + HH], gh = gb1[n + 2*HH];
    const float vz = vvec[n], vr = vvec[n + HH], vh = vvec[n + 2*HH];
    const float cz = cvec[n], cr = cvec[n + HH], ch = cvec[n + 2*HH];

#pragma unroll
    for (int t = 0; t < MT; ++t) {
        const int mbase = m0 + t * 16 + hi16 * 8;
#pragma unroll
        for (int e = 0; e < 8; ++e) {
            const int m = mbase + e;
            float col = h_in32[(size_t)m * HH + step] * invkm;
            float xz = fmaf(col, vz, cz);
            float xr = fmaf(col, vr, cr);
            float xh = fmaf(col, vh, ch);
            float z  = sigmoidf_(xz + acc[t][0][e] + gz);
            float r  = sigmoidf_(xr + acc[t][1][e] + gr);
            float hh = tanhf_(xh + r * (acc[t][2][e] + gh));
            float hold = h_in32[(size_t)m * HH + n];
            float hn = z * hold + (1.0f - z) * hh;
            h_out32[(size_t)m * HH + n] = hn;
            h_out16[(size_t)m * HH + n] = (_Float16)hn;
        }
    }
}

// Reference returns (h, h): duplicate into both halves of d_out.
__global__ void enc_copy_out(const float* __restrict__ h, float* __restrict__ out) {
    size_t idx = (size_t)blockIdx.x * 256 + threadIdx.x;
    float v = h[idx];
    out[idx] = v;
    out[idx + (size_t)BB * HH] = v;
}

extern "C" void kernel_launch(void* const* d_in, const int* in_sizes, int n_in,
                              void* d_out, int out_size, void* d_ws, size_t ws_size,
                              hipStream_t stream) {
    const float* x  = (const float*)d_in[0];   // (B, T)
    const float* km = (const float*)d_in[1];   // (T,)
    const float* dk = (const float*)d_in[2];   // (1, H)
    const float* db = (const float*)d_in[3];   // (H,)
    const float* Kk = (const float*)d_in[4];   // (H, 3H)
    const float* Rr = (const float*)d_in[5];   // (H, 3H)
    const float* gb = (const float*)d_in[6];   // (2, 3H)
    const float* gb0 = gb;
    const float* gb1 = gb + H3;
    float* out = (float*)d_out;

    // Workspace carve-up (~50 MB total)
    char* w = (char*)d_ws;
    float*    h32a = (float*)w;            w += (size_t)BB * HH * sizeof(float);
    float*    h32b = (float*)w;            w += (size_t)BB * HH * sizeof(float);
    _Float16* h16a = (_Float16*)w;         w += (size_t)BB * HH * sizeof(_Float16);
    _Float16* h16b = (_Float16*)w;         w += (size_t)BB * HH * sizeof(_Float16);
    _Float16* R16t = (_Float16*)w;         w += (size_t)H3 * HH * sizeof(_Float16);
    float*    vvec = (float*)w;            w += H3 * sizeof(float);
    float*    cvec = (float*)w;            w += H3 * sizeof(float);

    // One-time precompute per launch
    enc_prep_R<<<(H3 * HH) / 256, 256, 0, stream>>>(Rr, R16t);
    enc_prep_vc<<<(H3 + 255) / 256, 256, 0, stream>>>(dk, db, Kk, gb0, vvec, cvec);

    // Step 0 (h_prev = 0, elementwise)
    enc_step0<<<(BB * HH) / 256, 256, 0, stream>>>(x, km, vvec, cvec, gb1, h32a, h16a);

    // Steps 1..T-1: ping-pong between (a,b)
    float*    hs32[2] = { h32a, h32b };
    _Float16* hs16[2] = { h16a, h16b };
    int cur = 0;
    dim3 grid(BB / (16 * MT * 8), HH / 16);
    for (int i = 1; i < TT; ++i) {
        int nxt = 1 - cur;
        enc_step<<<grid, 256, 0, stream>>>(hs32[cur], hs16[cur], R16t, gb1,
                                           vvec, cvec, km, i,
                                           hs32[nxt], hs16[nxt]);
        cur = nxt;
    }

    enc_copy_out<<<(BB * HH) / 256, 256, 0, stream>>>(hs32[cur], out);
}
